// GaussianToBEV_81346680586531
// MI455X (gfx1250) — matrix operand order repaired and checked by reading, not compiled, not run on hardware
//
#include <hip/hip_runtime.h>

typedef __attribute__((ext_vector_type(16))) _Float16 v16h;
typedef __attribute__((ext_vector_type(8)))  float    v8f;

#define Bn   2
#define Hn   512
#define Wn   512
#define Dn   32
#define ZBn  16
#define HEn  32
#define Mn   1000000
#define Cn   64
#define HWn  (Hn * Wn)        // 262144 = 2^18
#define OUTC (Cn + HEn)       // 96

// ---------------------------------------------------------------------------
// Kernel 1: zero a region (uint4 granularity). Needed every call: atomics
// accumulate and the harness does not re-poison between replays.
// ---------------------------------------------------------------------------
__global__ __launch_bounds__(256) void zero4_kernel(uint4* __restrict__ p, unsigned n4) {
  unsigned i = blockIdx.x * 256u + threadIdx.x;
  if (i < n4) p[i] = make_uint4(0u, 0u, 0u, 0u);
}

// ---------------------------------------------------------------------------
// Kernel 2: scatter. One wave per point; lane handles 2 channels.
// Sums accumulate directly into the BEV region of d_out (channel-major), so
// the 64 dword atomics per point land on 64 different 1MB-strided lines ->
// good spread across L2 atomic channels. Feature reads are fully coalesced
// (float2 per lane = 256B per wave). Point index is forced wave-uniform so
// the coordinate load becomes a single scalar s_load_b128.
// Relaxed agent-scope atomics -> no-return GLOBAL_ATOMIC_ADD_F32/OR_B32,
// no fences; cross-kernel visibility comes from dispatch boundaries.
// ---------------------------------------------------------------------------
__global__ __launch_bounds__(256) void scatter_kernel(
    const float* __restrict__ feat, const int* __restrict__ vc,
    float* __restrict__ out, float* __restrict__ counts, unsigned* __restrict__ occ)
{
  const unsigned t    = blockIdx.x * 256u + threadIdx.x;
  const unsigned lane = t & 31u;
  const unsigned p    = __builtin_amdgcn_readfirstlane(t >> 5);  // wave-uniform
  if (p >= Mn) return;

  const int4 q = ((const int4*)vc)[p];      // [b, z, y, x] via scalar load
  const int bq = q.x, zq = q.y, yq = q.z, xq = q.w;
  const unsigned pin  = (unsigned)(yq * Wn + xq);
  const size_t   base = (size_t)bq * OUTC * HWn + pin;

  const float2 f = ((const float2*)(feat + (size_t)p * Cn))[lane];
  const unsigned c = lane * 2u;
  __hip_atomic_fetch_add(&out[base + (size_t)c        * HWn], f.x,
                         __ATOMIC_RELAXED, __HIP_MEMORY_SCOPE_AGENT);
  __hip_atomic_fetch_add(&out[base + (size_t)(c + 1u) * HWn], f.y,
                         __ATOMIC_RELAXED, __HIP_MEMORY_SCOPE_AGENT);

  if (lane == 0u) {
    const unsigned code = (unsigned)bq * HWn + pin;
    __hip_atomic_fetch_add(&counts[code], 1.0f,
                           __ATOMIC_RELAXED, __HIP_MEMORY_SCOPE_AGENT);
    int zb = zq >> 1;                       // z*ZB/D = z/2
    zb = zb < 0 ? 0 : (zb > 15 ? 15 : zb);
    __hip_atomic_fetch_or(&occ[code], 1u << zb,
                          __ATOMIC_RELAXED, __HIP_MEMORY_SCOPE_AGENT);
  }
}

// ---------------------------------------------------------------------------
// Kernel 3: finalize. One wave handles 16 consecutive pixels (same batch):
//   (a) BEV mean: divide 64 channels by max(count,1) in place.
//   (b) h_emb via v_wmma_f32_16x16x32_f16: M=16 pixels, K=16 occ bins
//       (padded to 32), N=16 embedding channels (x2 for HE=32).
// Grid divides exactly -> EXEC is all-ones at the WMMAs (ISA requirement).
// ---------------------------------------------------------------------------
__global__ __launch_bounds__(256) void finalize_kernel(
    const float* __restrict__ counts, const unsigned* __restrict__ occ,
    const float* __restrict__ Wh, const float* __restrict__ bh,
    float* __restrict__ out)
{
  const unsigned lane  = threadIdx.x & 31u;
  const unsigned wave  = threadIdx.x >> 5;
  const unsigned group = blockIdx.x * 8u + wave;   // 32768 groups total
  const unsigned pixg  = group << 4;               // global pixel base
  const unsigned bb    = pixg >> 18;               // / HW
  const unsigned pin   = pixg & (HWn - 1u);        // pixel-in-batch base
  const unsigned px    = lane & 15u;               // pixel within group
  const unsigned hi    = lane >> 4;                // lane half

  // ---- (a) BEV normalize in place ----
  const float cnt = counts[pixg + px];
  const float inv = 1.0f / fmaxf(cnt, 1.0f);
  const size_t obase = (size_t)bb * OUTC * HWn + pin + px;
  #pragma unroll
  for (int k = 0; k < 16; ++k) {
    const unsigned cl = 4u * k + 2u * hi;          // half-wave splits channels
    const size_t o0 = obase + (size_t)cl * HWn;
    out[o0]       *= inv;
    out[o0 + HWn] *= inv;
  }

  // ---- (b) h_emb = occ(16x16pad32) x Wh^T(32x16) + bh, via WMMA ----
  // A layout (16-bit A 16x32, ISA 7.12.2): lanes0-15 row M=px hold K=0..7 in
  // halfs 0..7 (and K=16..23 in 8..15); lanes16-31 hold K=8..15 (and 24..31).
  const unsigned bits = occ[pixg + px];
  const unsigned sh   = hi * 8u;
  v16h A;
  #pragma unroll
  for (int i = 0; i < 8; ++i)
    A[i] = (_Float16)(float)((bits >> (sh + i)) & 1u);
  #pragma unroll
  for (int i = 8; i < 16; ++i)
    A[i] = (_Float16)0.0f;                          // K>=16 padding

  // B layout (16-bit B striping, derived from ISA 7.12.4 B pattern):
  // lanes0-15 (col N=px) carry K=0..7 = Wh[n][0..7] in halfs 0..7; lanes16-31
  // carry K=8..15 = Wh[n][8..15]; halfs 8..15 (K=16..31) are zero padding.
  const float* w0 = Wh + (size_t)px * ZBn;          // rows 0..15  (N base 0)
  const float* w1 = Wh + (size_t)(px + 16u) * ZBn;  // rows 16..31 (N base 16)
  v16h B0, B1;
  #pragma unroll
  for (int i = 0; i < 16; ++i) {
    B0[i] = (i >= 8) ? (_Float16)0.0f : (_Float16)w0[hi * 8u + i];
    B1[i] = (i >= 8) ? (_Float16)0.0f : (_Float16)w1[hi * 8u + i];
  }

  // C = bias broadcast: D[m][n] starts at bh[nbase+n] for all m.
  const float bias0 = bh[px];
  const float bias1 = bh[px + 16u];
  v8f C0, C1;
  #pragma unroll
  for (int r = 0; r < 8; ++r) { C0[r] = bias0; C1[r] = bias1; }

  v8f D0 = __builtin_amdgcn_wmma_f32_16x16x32_f16(false, A, false, B0,
                                                  (short)0, C0, false, false);
  v8f D1 = __builtin_amdgcn_wmma_f32_16x16x32_f16(false, A, false, B1,
                                                  (short)0, C1, false, false);

  // D layout (32-bit C/D 16x16): VGPR r, lanes0-15 -> M=r, N=lane;
  // lanes16-31 -> M=r+8, N=lane-16.  channel = 64 + nbase + N, pixel = M.
  const size_t oh = ((size_t)bb * OUTC + Cn + px) * HWn + pin + hi * 8u;
  #pragma unroll
  for (int r = 0; r < 8; ++r) {
    out[oh + r]                    = D0[r];   // channels 64..79
    out[oh + (size_t)16 * HWn + r] = D1[r];   // channels 80..95
  }
}

// ---------------------------------------------------------------------------
extern "C" void kernel_launch(void* const* d_in, const int* in_sizes, int n_in,
                              void* d_out, int out_size, void* d_ws, size_t ws_size,
                              hipStream_t stream)
{
  const float* feat = (const float*)d_in[0];   // (M, 64) f32
  const int*   vc   = (const int*)  d_in[1];   // (M, 4)  i32
  const float* Wh   = (const float*)d_in[2];   // (32, 16) f32
  const float* bh   = (const float*)d_in[3];   // (32,)    f32
  float* out = (float*)d_out;                  // (2, 96, 512, 512) f32

  float*    counts = (float*)d_ws;                       // B*HW floats
  unsigned* occ    = (unsigned*)d_ws + (size_t)Bn * HWn; // B*HW u32  (4MB total)

  // Zero the BEV accumulation regions of out (channels 0..63 of each batch)
  // and the counts/occ workspace.
  {
    const unsigned n4 = (Cn * HWn) / 4;                  // 4,194,304 uint4 per batch
    zero4_kernel<<<n4 / 256, 256, 0, stream>>>((uint4*)out, n4);
    zero4_kernel<<<n4 / 256, 256, 0, stream>>>(
        (uint4*)(out + (size_t)OUTC * HWn), n4);
    const unsigned w4 = (2u * Bn * HWn) / 4;             // 262,144 uint4
    zero4_kernel<<<w4 / 256, 256, 0, stream>>>((uint4*)d_ws, w4);
  }

  // Scatter: one wave per point.
  scatter_kernel<<<(Mn * 32) / 256, 256, 0, stream>>>(feat, vc, out, counts, occ);

  // Finalize: 32768 pixel-groups, 8 waves/block.
  finalize_kernel<<<(Bn * HWn / 16) / 8, 256, 0, stream>>>(counts, occ, Wh, bh, out);
}
